// TemporalDecoder_80848464380112
// MI455X (gfx1250) — matrix operand order repaired and checked by reading, not compiled, not run on hardware
//
#include <hip/hip_runtime.h>
#include <hip/hip_bf16.h>

// ---------------------------------------------------------------------------
// 2-layer LSTM (H=256) over BN=1600 independent rows, 64 encoder + 10 decoder
// steps, layernorm + 3 MLP heads.  Per-row recurrence => one persistent
// workgroup per 16-row tile, no inter-block sync.  bf16 WMMA (f32 accum) for
// gate GEMMs; weights pre-swizzled to bf16 B-fragments streamed from L2.
// Gate accumulators AND cell state live in registers: each wave owns 2 hidden
// column tiles and computes all four gate tiles (i,f,g,o) for those columns,
// so the cell update is lane-local.  K loop stays rolled (bounded scheduling
// window -> no spills); weight addresses are laundered via an opaque integer
// offset so loads stream from L2 as global_load (not cached across steps,
// and not demoted to flat_load).
// ---------------------------------------------------------------------------
#define HDIM   256
#define G4     1024
#define SLEN   64
#define NNODE  100
#define BN     1600
#define ROWS   16
#define TPB    256           // 8 wave32s; 8 waves * 2 tiles = 16 hidden tiles
#define NBLOCKS (BN / ROWS)  // 100
#define WFRAG_ELEMS 512
#define WMAT_ELEMS  (64 * 8 * WFRAG_ELEMS)  // 262144 elems per 1024x256 matrix

typedef __attribute__((ext_vector_type(16))) __bf16 bf16x16;
typedef __attribute__((ext_vector_type(8)))  __bf16 bf16x8;
typedef __attribute__((ext_vector_type(8)))  float  f32x8;

__device__ __forceinline__ unsigned short f32_to_bf16(float f) {
  unsigned int u = __builtin_bit_cast(unsigned int, f);
  u += 0x7FFFu + ((u >> 16) & 1u);   // round-to-nearest-even
  return (unsigned short)(u >> 16);
}

// Native gfx1250 v_tanh_f32 if available (TRANS32 op; co-issues with XDL).
#if __has_builtin(__builtin_amdgcn_tanhf)
__device__ __forceinline__ float fast_tanh(float x) {
  return __builtin_amdgcn_tanhf(x);
}
#elif __has_builtin(__builtin_amdgcn_tanh_f32)
__device__ __forceinline__ float fast_tanh(float x) {
  return __builtin_amdgcn_tanh_f32(x);
}
#else
__device__ __forceinline__ float fast_tanh(float x) { return tanhf(x); }
#endif
__device__ __forceinline__ float fast_sigmoid(float x) {
  return 0.5f * fast_tanh(0.5f * x) + 0.5f;   // branch-free
}

// ---------------------------------------------------------------------------
// Prep: fp32 weights -> bf16 WMMA B-fragments.  Fragment(nt,kb) = 32(K)x16(N)
// block of W^T stored [half(2)][lane(32)][8 bf16] => two coalesced 16B loads
// per lane (1KB contiguous per wave) in the hot loop.
// ---------------------------------------------------------------------------
__global__ void swizzle_weights(const float* __restrict__ W0,
                                const float* __restrict__ W1,
                                const float* __restrict__ W2,
                                const float* __restrict__ W3,
                                unsigned short* __restrict__ out) {
  int o = blockIdx.x * blockDim.x + threadIdx.x;
  if (o >= 4 * WMAT_ELEMS) return;
  int mat  = o >> 18;
  int r    = o & (WMAT_ELEMS - 1);
  int frag = r >> 9;            // nt*8 + kb
  int nt   = frag >> 3;
  int kb   = frag & 7;
  int w    = r & 511;
  int half = w >> 8;
  int lane = (w >> 3) & 31;
  int j    = w & 7;
  int row  = nt * 16 + (lane & 15);
  int k    = kb * 32 + ((lane >> 4) << 3) + (half << 4) + j;
  const float* src = (mat == 0) ? W0 : (mat == 1) ? W1 : (mat == 2) ? W2 : W3;
  out[o] = f32_to_bf16(src[row * HDIM + k]);
}

// B fragment: 2 coalesced 16B global loads (L2-resident weights).
__device__ __forceinline__ bf16x16 load_frag_b(const unsigned short* frag,
                                               int lane) {
  bf16x8 b0 = *(const bf16x8*)(frag + (lane << 3));
  bf16x8 b1 = *(const bf16x8*)(frag + 256 + (lane << 3));
  return __builtin_shufflevector(b0, b1,
      0,1,2,3,4,5,6,7,8,9,10,11,12,13,14,15);
}
// A fragment from LDS row-major bf16 [16][256]; documented 16-bit A layout
// (lanes 0-15: K 0-7 & 16-23; lanes 16-31: K 8-15 & 24-31 per 32-K block).
__device__ __forceinline__ bf16x16 load_frag_a(const unsigned short* A,
                                               int arow, int k0) {
  bf16x8 a0 = *(const bf16x8*)(A + arow + k0);
  bf16x8 a1 = *(const bf16x8*)(A + arow + k0 + 16);
  return __builtin_shufflevector(a0, a1,
      0,1,2,3,4,5,6,7,8,9,10,11,12,13,14,15);
}

// ---------------------------------------------------------------------------
// One LSTM cell, fully register-resident update.
// Wave `wv` owns hidden tiles nt = 2*wv, 2*wv+1.  Gate g's WMMA tile index is
// g*16+nt, so i/f/g/o accumulators for a (row,col) element land in the same
// lane (D-layout: lane = (m>=8)*16 + n, VGPR j -> m = j + 8*(lane>=16)).
// c stays in f32x8 registers; new h goes to a DIFFERENT buffer than h_prev
// (ping-pong), so one barrier per cell suffices (caller's).
// ---------------------------------------------------------------------------
__device__ __forceinline__ void lstm_cell_reg(
    const unsigned short* __restrict__ Ax,     // LDS bf16 [16][256] input
    const unsigned short* __restrict__ AhIn,   // LDS bf16 h_prev
    unsigned short* __restrict__ AhOut,        // LDS bf16 h_new (!= AhIn)
    float* __restrict__ hF32,                  // optional LDS f32 h_new
    const unsigned short* __restrict__ Wih,    // swizzled fragment base
    const unsigned short* __restrict__ Whh,
    const float (&bias)[2][4],                 // combined bih+bhh per tile/gate
    f32x8 (&cr)[2],                            // cell state registers
    int tid)
{
  // Opaque zero offset: addresses are formally loop-variant, so weight loads
  // cannot be CSE'd/hoisted across timesteps (must stream from L2), while
  // the pointer bases keep global address-space provenance (global_load).
  long woff = 0;
  asm volatile("" : "+s"(woff));
  const unsigned short* wih = Wih + woff;
  const unsigned short* whh = Whh + woff;

  const int lane = tid & 31;
  const int wave = tid >> 5;
  const int nlo  = lane & 15;
  const int hi   = lane >> 4;
  const int arow = nlo * HDIM;
#pragma unroll
  for (int t2 = 0; t2 < 2; ++t2) {
    const int nt = wave * 2 + t2;              // hidden col tile 0..15
    f32x8 acc[4] = {{}, {}, {}, {}};           // i, f, g, o
#pragma unroll 1
    for (int kb = 0; kb < 8; ++kb) {           // rolled: bounds the scheduling
      const int k0 = kb * 32 + hi * 8;         // window -> no spills
      bf16x16 Xv = load_frag_a(Ax,   arow, k0);
      bf16x16 Hv = load_frag_a(AhIn, arow, k0);
#pragma unroll
      for (int g = 0; g < 4; ++g) {
        const int fidx = ((g * 16 + nt) * 8 + kb) << 9;
        bf16x16 Bi = load_frag_b(wih + fidx, lane);
        acc[g] = __builtin_amdgcn_wmma_f32_16x16x32_bf16(
            false, Xv, false, Bi, (short)0, acc[g], false, false);
        bf16x16 Bh = load_frag_b(whh + fidx, lane);
        acc[g] = __builtin_amdgcn_wmma_f32_16x16x32_bf16(
            false, Hv, false, Bh, (short)0, acc[g], false, false);
      }
    }
    // Lane-local gate nonlinearity + state update (torch order i,f,g,o).
    const int col = nt * 16 + nlo;
#pragma unroll
    for (int j = 0; j < 8; ++j) {
      float gi = acc[0][j] + bias[t2][0];
      float gf = acc[1][j] + bias[t2][1];
      float gg = acc[2][j] + bias[t2][2];
      float go = acc[3][j] + bias[t2][3];
      float si = fast_sigmoid(gi);
      float sf = fast_sigmoid(gf);
      float so = fast_sigmoid(go);
      float cv = sf * cr[t2][j] + si * fast_tanh(gg);
      cr[t2][j] = cv;
      float hn = so * fast_tanh(cv);
      const int m = hi * 8 + j;
      AhOut[m * HDIM + col] = f32_to_bf16(hn);
      if (hF32) hF32[m * HDIM + col] = hn;
    }
  }
}

// Layernorm over H=256 per row: 16 threads/row, shfl_xor reduction width 16.
__device__ __forceinline__ void layernorm_rows(
    const float* __restrict__ h, const float* __restrict__ g,
    const float* __restrict__ b, float* __restrict__ outf,
    unsigned short* __restrict__ outbf, int tid)
{
  const int r = tid >> 4, lp = tid & 15;
  float s = 0.f, s2 = 0.f;
  for (int j = 0; j < 16; ++j) {
    float v = h[r * HDIM + lp * 16 + j];
    s += v; s2 += v * v;
  }
  for (int m = 8; m >= 1; m >>= 1) {
    s  += __shfl_xor(s,  m, 16);
    s2 += __shfl_xor(s2, m, 16);
  }
  float mean = s * (1.f / HDIM);
  float var  = s2 * (1.f / HDIM) - mean * mean;
  float inv  = rsqrtf(var + 1e-5f);
  for (int j = 0; j < 16; ++j) {
    int k = lp * 16 + j;
    float v = (h[r * HDIM + k] - mean) * inv * g[k] + b[k];
    outf[r * HDIM + k] = v;
    outbf[r * HDIM + k] = f32_to_bf16(v);
  }
}

// Heads: cen(256->128->4), svd(256->128->2), lsvd(256->256->16) => 22 outputs.
__device__ __forceinline__ void decoder_heads(
    const float* __restrict__ htf, float* __restrict__ scratch,
    const float* cw1, const float* cb1, const float* cw2, const float* cb2,
    const float* sw1, const float* sb1, const float* sw2, const float* sb2,
    const float* lw1, const float* lb1, const float* lw2, const float* lb2,
    float* __restrict__ out, int rowbase, int step, int steps, int tid)
{
  float* cenh = scratch;          // [16][128]
  float* svdh = scratch + 2048;   // [16][128]
  float* lsvh = scratch + 4096;   // [16][256]
  for (int idx = tid; idx < ROWS * 128; idx += TPB) {
    int m = idx >> 7, j = idx & 127;
    const float* xr = htf + m * HDIM;
    const float* w = cw1 + j * HDIM;
    const float* v = sw1 + j * HDIM;
    float a = cb1[j], bb = sb1[j];
    for (int k = 0; k < HDIM; ++k) { a += xr[k] * w[k]; bb += xr[k] * v[k]; }
    cenh[idx] = fmaxf(a, 0.f);
    svdh[idx] = fmaxf(bb, 0.f);
  }
  for (int idx = tid; idx < ROWS * HDIM; idx += TPB) {
    int m = idx >> 8, j = idx & 255;
    const float* xr = htf + m * HDIM;
    const float* w = lw1 + j * HDIM;
    float a = lb1[j];
    for (int k = 0; k < HDIM; ++k) a += xr[k] * w[k];
    lsvh[idx] = fmaxf(a, 0.f);
  }
  __syncthreads();
  for (int idx = tid; idx < ROWS * 22; idx += TPB) {
    int m = idx / 22, j = idx % 22;
    float a;
    if (j < 4) {
      const float* w = cw2 + j * 128; const float* xh = cenh + m * 128;
      a = cb2[j];
      for (int k = 0; k < 128; ++k) a += xh[k] * w[k];
    } else if (j < 6) {
      int jj = j - 4;
      const float* w = sw2 + jj * 128; const float* xh = svdh + m * 128;
      a = sb2[jj];
      for (int k = 0; k < 128; ++k) a += xh[k] * w[k];
    } else {
      int jj = j - 6;
      const float* w = lw2 + jj * HDIM; const float* xh = lsvh + m * HDIM;
      a = lb2[jj];
      for (int k = 0; k < HDIM; ++k) a += xh[k] * w[k];
    }
    int bn = rowbase + m;
    int b = bn / NNODE, n = bn % NNODE;
    out[(((size_t)b * steps + step) * NNODE + n) * 22 + j] = a;
  }
}

// ---------------------------------------------------------------------------
// Persistent kernel.  LDS: h1f/htf f32 (32KB) + heads scratch (32KB) +
// 5 bf16 [16][256] buffers (Ain + ping-pong h0/h1) (40KB) = ~104KB < 320KB.
// Cell state c0/c1 and gate accumulators never leave VGPRs.
// ---------------------------------------------------------------------------
__global__ void __launch_bounds__(TPB)
lstm_decoder_kernel(
    const float* __restrict__ x,
    const unsigned short* __restrict__ Wsw,
    const float* __restrict__ bih0, const float* __restrict__ bhh0,
    const float* __restrict__ bih1, const float* __restrict__ bhh1,
    const float* __restrict__ ln_g, const float* __restrict__ ln_b,
    const float* cw1, const float* cb1, const float* cw2, const float* cb2,
    const float* sw1, const float* sb1, const float* sw2, const float* sb2,
    const float* lw1, const float* lb1, const float* lw2, const float* lb2,
    const int* __restrict__ steps_p, float* __restrict__ out)
{
  extern __shared__ __align__(16) char smem[];
  float* h1f      = (float*)smem;                  // [16][256] f32
  float* htf      = h1f + ROWS * HDIM;             // [16][256] f32
  float* hscratch = htf + ROWS * HDIM;             // 8192 f32 (heads)
  unsigned short* Ain  = (unsigned short*)(hscratch + 8192);
  unsigned short* Ah0a = Ain  + ROWS * HDIM;
  unsigned short* Ah0b = Ah0a + ROWS * HDIM;
  unsigned short* Ah1a = Ah0b + ROWS * HDIM;
  unsigned short* Ah1b = Ah1a + ROWS * HDIM;
  unsigned short* h0buf[2] = {Ah0a, Ah0b};
  unsigned short* h1buf[2] = {Ah1a, Ah1b};

  const int tid = threadIdx.x;
  const int rowbase = blockIdx.x * ROWS;
  const int steps = steps_p[0];
  const int lane = tid & 31, wave = tid >> 5, nlo = lane & 15;

  const unsigned short* Wih0 = Wsw;
  const unsigned short* Whh0 = Wsw + WMAT_ELEMS;
  const unsigned short* Wih1 = Wsw + 2 * WMAT_ELEMS;
  const unsigned short* Whh1 = Wsw + 3 * WMAT_ELEMS;

  // Combined biases for this lane's columns (fixed for whole run).
  float bias0[2][4], bias1[2][4];
#pragma unroll
  for (int t2 = 0; t2 < 2; ++t2) {
    const int col = (wave * 2 + t2) * 16 + nlo;
#pragma unroll
    for (int g = 0; g < 4; ++g) {
      bias0[t2][g] = bih0[g * HDIM + col] + bhh0[g * HDIM + col];
      bias1[t2][g] = bih1[g * HDIM + col] + bhh1[g * HDIM + col];
    }
  }
  f32x8 cr0[2] = {{}, {}};
  f32x8 cr1[2] = {{}, {}};

  for (int idx = tid; idx < ROWS * HDIM; idx += TPB) {
    Ah0a[idx] = 0; Ah1a[idx] = 0;   // h_prev = 0 at t=0 (p starts at 0)
  }
  __syncthreads();

  int p = 0;
  // ---- encoder: 64 steps ----
  for (int s = 0; s < SLEN; ++s) {
    for (int idx = tid; idx < ROWS * HDIM; idx += TPB) {
      int m = idx >> 8, k = idx & 255;
      int bn = rowbase + m;
      int b = bn / NNODE, n = bn % NNODE;
      Ain[idx] = f32_to_bf16(x[(((size_t)b * SLEN + s) * NNODE + n) * HDIM + k]);
    }
    __syncthreads();
    lstm_cell_reg(Ain, h0buf[p], h0buf[p ^ 1], nullptr,
                  Wih0, Whh0, bias0, cr0, tid);
    __syncthreads();
    lstm_cell_reg(h0buf[p ^ 1], h1buf[p], h1buf[p ^ 1], h1f,
                  Wih1, Whh1, bias1, cr1, tid);
    __syncthreads();
    p ^= 1;
  }
  layernorm_rows(h1f, ln_g, ln_b, htf, Ain, tid);
  __syncthreads();

  // ---- decoder ----
  for (int st = 0; st < steps; ++st) {
    decoder_heads(htf, hscratch, cw1, cb1, cw2, cb2, sw1, sb1, sw2, sb2,
                  lw1, lb1, lw2, lb2, out, rowbase, st, steps, tid);
    __syncthreads();
    lstm_cell_reg(Ain, h0buf[p], h0buf[p ^ 1], nullptr,
                  Wih0, Whh0, bias0, cr0, tid);
    __syncthreads();
    lstm_cell_reg(h0buf[p ^ 1], h1buf[p], h1buf[p ^ 1], h1f,
                  Wih1, Whh1, bias1, cr1, tid);
    __syncthreads();
    layernorm_rows(h1f, ln_g, ln_b, htf, Ain, tid);
    __syncthreads();
    p ^= 1;
  }
}

extern "C" void kernel_launch(void* const* d_in, const int* in_sizes, int n_in,
                              void* d_out, int out_size, void* d_ws, size_t ws_size,
                              hipStream_t stream) {
  const float* x     = (const float*)d_in[0];
  const float* W_ih0 = (const float*)d_in[1];
  const float* W_hh0 = (const float*)d_in[2];
  const float* b_ih0 = (const float*)d_in[3];
  const float* b_hh0 = (const float*)d_in[4];
  const float* W_ih1 = (const float*)d_in[5];
  const float* W_hh1 = (const float*)d_in[6];
  const float* b_ih1 = (const float*)d_in[7];
  const float* b_hh1 = (const float*)d_in[8];
  const float* ln_g  = (const float*)d_in[9];
  const float* ln_b  = (const float*)d_in[10];
  const float* cw1   = (const float*)d_in[11];
  const float* cb1   = (const float*)d_in[12];
  const float* cw2   = (const float*)d_in[13];
  const float* cb2   = (const float*)d_in[14];
  const float* sw1   = (const float*)d_in[15];
  const float* sb1   = (const float*)d_in[16];
  const float* sw2   = (const float*)d_in[17];
  const float* sb2   = (const float*)d_in[18];
  const float* lw1   = (const float*)d_in[19];
  const float* lb1   = (const float*)d_in[20];
  const float* lw2   = (const float*)d_in[21];
  const float* lb2   = (const float*)d_in[22];
  const int* steps_p = (const int*)d_in[23];
  float* out = (float*)d_out;
  unsigned short* Wsw = (unsigned short*)d_ws;  // 2MB bf16 weight fragments

  const int total = 4 * WMAT_ELEMS;
  swizzle_weights<<<(total + TPB - 1) / TPB, TPB, 0, stream>>>(
      W_ih0, W_hh0, W_ih1, W_hh1, Wsw);

  const size_t smem_bytes =
      (size_t)(2 * ROWS * HDIM + 8192) * sizeof(float) +
      (size_t)(5 * ROWS * HDIM) * sizeof(unsigned short);
  lstm_decoder_kernel<<<NBLOCKS, TPB, smem_bytes, stream>>>(
      x, Wsw, b_ih0, b_hh0, b_ih1, b_hh1, ln_g, ln_b,
      cw1, cb1, cw2, cb2, sw1, sb1, sw2, sb2, lw1, lb1, lw2, lb2,
      steps_p, out);
}
